// CMCV2Loss_2508260900944
// MI455X (gfx1250) — matrix operand order repaired and checked by reading, not compiled, not run on hardware
//
#include <hip/hip_runtime.h>
#include <hip/hip_bf16.h>

typedef __attribute__((ext_vector_type(16))) __bf16 v16bf;
typedef __attribute__((ext_vector_type(8)))  __bf16 bf16x8;
typedef __attribute__((ext_vector_type(4)))  __bf16 bf16x4;
typedef __attribute__((ext_vector_type(8)))  float  v8f;

#define B_ROWS 2048
#define DHALF  512
#define NTOT   4096      // 2*B
#define INV_T  5.0f      // 1/0.2
#define EPS_N  1e-8f
#define NLOSS  9
#define ZN_ELEMS (12UL * B_ROWS * DHALF)

// -------------------- zero output + row-sum accumulators --------------------
__global__ void zero_kernel(float* out, float* rowS) {
    int idx = blockIdx.x * 256 + threadIdx.x;
    if (idx < NLOSS * NTOT) rowS[idx] = 0.0f;
    if (idx == 0) out[0] = 0.0f;
}

// -------------------- row-normalize halves to bf16 --------------------
__global__ void row_normalize_kernel(const float* p0, const float* p1,
                                     const float* p2, const float* p3,
                                     const float* p4, const float* p5,
                                     __bf16* zn) {
    int gb = blockIdx.x;
    int i  = gb >> 11;       // input index 0..5
    int r  = gb & 2047;      // row
    const float* srcs[6] = {p0, p1, p2, p3, p4, p5};
    const float* src = srcs[i] + (size_t)r * 1024;

    int t = threadIdx.x;     // 0..255
    float x0 = src[t],        x1 = src[t + 256];   // shared half cols
    float y0 = src[t + 512],  y1 = src[t + 768];   // private half cols
    float ssh = x0 * x0 + x1 * x1;
    float spr = y0 * y0 + y1 * y1;
    #pragma unroll
    for (int m = 16; m >= 1; m >>= 1) {
        ssh += __shfl_xor(ssh, m);
        spr += __shfl_xor(spr, m);
    }
    __shared__ float rsh[8], rpr[8];
    int wid = t >> 5, lane = t & 31;
    if (lane == 0) { rsh[wid] = ssh; rpr[wid] = spr; }
    __syncthreads();
    if (t == 0) {
        float a = 0.f, b = 0.f;
        #pragma unroll
        for (int w = 0; w < 8; ++w) { a += rsh[w]; b += rpr[w]; }
        rsh[0] = fmaxf(sqrtf(a), EPS_N);
        rpr[0] = fmaxf(sqrtf(b), EPS_N);
    }
    __syncthreads();
    float ns = rsh[0], np = rpr[0];

    __bf16* osh = zn + (size_t)i       * B_ROWS * DHALF + (size_t)r * DHALF;
    __bf16* opr = zn + (size_t)(6 + i) * B_ROWS * DHALF + (size_t)r * DHALF;
    osh[t]       = (__bf16)(x0 / ns);
    osh[t + 256] = (__bf16)(x1 / ns);
    opr[t]       = (__bf16)(y0 / np);
    opr[t + 256] = (__bf16)(y1 / np);
}

// -------------------- 12 ortho (1 - cos) terms, fp32 --------------------
__global__ void ortho_kernel(const float* p0, const float* p1, const float* p2,
                             const float* p3, const float* p4, const float* p5,
                             float* out) {
    const float* srcs[6] = {p0, p1, p2, p3, p4, p5};
    const int pi_[3] = {0, 0, 1};
    const int pj_[3] = {1, 2, 2};

    int term = blockIdx.y;        // 0..11
    int view = term / 6;
    int tt   = term % 6;
    int base = view * 3;
    int wid  = threadIdx.x >> 5, lane = threadIdx.x & 31;
    int r    = blockIdx.x * 8 + wid;

    const float *x1, *x2;
    if (tt < 3) {  // shared_i vs private_i
        x1 = srcs[base + tt] + (size_t)r * 1024;
        x2 = x1 + 512;
    } else {       // private_i vs private_j
        x1 = srcs[base + pi_[tt - 3]] + (size_t)r * 1024 + 512;
        x2 = srcs[base + pj_[tt - 3]] + (size_t)r * 1024 + 512;
    }
    float dot = 0.f, na = 0.f, nb = 0.f;
    #pragma unroll 4
    for (int e = lane; e < 512; e += 32) {
        float a = x1[e], b = x2[e];
        dot += a * b; na += a * a; nb += b * b;
    }
    #pragma unroll
    for (int m = 16; m >= 1; m >>= 1) {
        dot += __shfl_xor(dot, m);
        na  += __shfl_xor(na,  m);
        nb  += __shfl_xor(nb,  m);
    }
    if (lane == 0) {
        float c = dot / (fmaxf(sqrtf(na), EPS_N) * fmaxf(sqrtf(nb), EPS_N));
        atomicAdd(out, (1.0f - c) * (1.0f / 2048.0f));
    }
}

// -------------------- 9 NT-Xent sims: triangular WMMA pass --------------------
// Exploits sim symmetry: block i owns rows [128i,128i+128) and computes col
// tiles ct in [8i,256). Strictly-upper tiles (ct >= 8i+8) contribute twice:
// row-side into per-wave registers, col-side via per-lane v-sums into an LDS
// colacc[] (ds_add_f32), flushed to global rowS with fp32 atomics. Blocks are
// paired (i, 31-i) so every block does exactly 264 tiles. Positives (r, r+2048)
// are always strictly-upper for r<2048 and serve row r+2048 by symmetry.
__global__ void __launch_bounds__(256)
ntxent_kernel(const __bf16* __restrict__ zn, float* rowS, float* posv) {
    const int aidx[9] = {0, 0, 1, 3, 3, 4, 6, 7, 8};
    const int bidx[9] = {1, 2, 2, 4, 5, 5, 9, 10, 11};
    int l = blockIdx.y;
    const __bf16* matA = zn + (size_t)aidx[l] * B_ROWS * DHALF;
    const __bf16* matB = zn + (size_t)bidx[l] * B_ROWS * DHALF;
    float* rowSl = rowS + (size_t)l * NTOT;
    float* posvl = posv + (size_t)l * B_ROWS;

    int tid  = threadIdx.x;
    int wid  = tid >> 5, lane = tid & 31;

    __shared__ __align__(16) __bf16 Bt[2][DHALF * 16];  // 2 x 16KB ping-pong
    __shared__ float colacc[NTOT];                      // 16KB col-side sums

    // staging assignment: thread -> (4-col group nn0, 8-K chunk k8)
    int nn0 = (tid & 3) * 4;      // 0,4,8,12
    int k8  = tid >> 2;           // 0..63

    auto colrow = [&](int c) -> const __bf16* {
        return (c < B_ROWS) ? (matA + (size_t)c * DHALF)
                            : (matB + (size_t)(c - B_ROWS) * DHALF);
    };
    auto stage_load = [&](int c0, bf16x8& r0, bf16x8& r1, bf16x8& r2, bf16x8& r3) {
        int c = c0 + nn0;
        r0 = *(const bf16x8*)(colrow(c + 0) + k8 * 8);
        r1 = *(const bf16x8*)(colrow(c + 1) + k8 * 8);
        r2 = *(const bf16x8*)(colrow(c + 2) + k8 * 8);
        r3 = *(const bf16x8*)(colrow(c + 3) + k8 * 8);
    };
    auto stage_store = [&](int b, bf16x8 r0, bf16x8 r1, bf16x8 r2, bf16x8 r3) {
        #pragma unroll
        for (int k = 0; k < 8; ++k) {
            bf16x4 w = {r0[k], r1[k], r2[k], r3[k]};
            *(bf16x4*)&Bt[b][(k8 * 8 + k) * 16 + nn0] = w;  // ds_store_b64
        }
    };
    auto bfrag = [&](int b, int k) -> v16bf {  // ISA B layout: lane=N, K in table order
        int kb = k * 32 + ((lane < 16) ? 0 : 8);
        v16bf f;
        #pragma unroll
        for (int s = 0; s < 16; ++s)
            f[s] = Bt[b][(kb + (s < 8 ? s : s + 8)) * 16 + (lane & 15)];
        return f;
    };

    int n     = lane & 15;            // D-matrix: lane holds column n
    int rowlo = (lane < 16) ? 0 : 8;  // D-matrix: VGPR v holds row v+rowlo

    for (int half = 0; half < 2; ++half) {
        int i = (half == 0) ? (int)blockIdx.x : 31 - (int)blockIdx.x;
        int ctstart = 8 * i;          // first col tile (band-diagonal start)
        int zbase = i * 128 + wid * 16;

        // ---- resident A fragments (ISA 16-bit A layout) ----
        int zr = zbase + (lane & 15);
        const __bf16* arow = (zr < B_ROWS) ? (matA + (size_t)zr * DHALF)
                                           : (matB + (size_t)(zr - B_ROWS) * DHALF);
        int koff = (lane < 16) ? 0 : 8;
        v16bf afrag[16];
        #pragma unroll
        for (int k = 0; k < 16; ++k) {
            bf16x8 lo = *(const bf16x8*)(arow + k * 32 + koff);
            bf16x8 hi = *(const bf16x8*)(arow + k * 32 + koff + 16);
            afrag[k] = __builtin_shufflevector(lo, hi, 0, 1, 2, 3, 4, 5, 6, 7,
                                                       8, 9, 10, 11, 12, 13, 14, 15);
        }

        float rowsum[8], dexp[8], plog[8];
        #pragma unroll
        for (int v = 0; v < 8; ++v) { rowsum[v] = 0.f; dexp[v] = 0.f; plog[v] = 0.f; }

        // init col-side accumulator + stage first tile
        for (int j = tid; j < NTOT; j += 256) colacc[j] = 0.f;
        {
            bf16x8 r0, r1, r2, r3;
            stage_load(ctstart * 16, r0, r1, r2, r3);
            stage_store(0, r0, r1, r2, r3);
        }
        __syncthreads();

        for (int ct = ctstart; ct < NTOT / 16; ++ct) {
            int p  = (ct - ctstart) & 1;
            int c0 = ct * 16;

            // 16x16 output tile: 16 WMMAs over K=512
            v8f acc = {0.f, 0.f, 0.f, 0.f, 0.f, 0.f, 0.f, 0.f};
            #pragma unroll
            for (int k = 0; k < 16; ++k) {
                acc = __builtin_amdgcn_wmma_f32_16x16x32_bf16(
                    false, afrag[k], false, bfrag(p, k), (short)0, acc, false, false);
            }

            // fused streaming softmax pieces (logits <= ~5, exp safe)
            float cs = 0.f;   // col-side partial: sum over this wave's 16 rows
            #pragma unroll
            for (int v = 0; v < 8; ++v) {
                int row  = zbase + v + rowlo;
                int col  = c0 + n;
                float s  = acc[v] * INV_T;
                float e  = __expf(s);
                rowsum[v] += e;
                cs        += e;
                if (col == row) dexp[v] += e;             // self-similarity
                int posc = (row + B_ROWS) & (NTOT - 1);
                if (col == posc) plog[v] += s;            // positive logit
            }
            // col-side: only strictly-upper tiles (avoid double count in band)
            if (ct >= ctstart + 8) {
                cs += __shfl_xor(cs, 16);                 // both row halves
                if (lane < 16) atomicAdd(&colacc[c0 + n], cs);  // ds_add_f32
            }

            // stage next tile into the other buffer; one barrier per tile
            if (ct + 1 < NTOT / 16) {
                bf16x8 r0, r1, r2, r3;
                stage_load(c0 + 16, r0, r1, r2, r3);
                stage_store(1 - p, r0, r1, r2, r3);
            }
            __syncthreads();
        }

        // ---- row-side writeback: reduce across the 16 lanes per row ----
        #pragma unroll
        for (int v = 0; v < 8; ++v) {
            float S = rowsum[v], de = dexp[v], pl = plog[v];
            #pragma unroll
            for (int msk = 1; msk < 16; msk <<= 1) {
                S  += __shfl_xor(S,  msk);
                de += __shfl_xor(de, msk);
                pl += __shfl_xor(pl, msk);
            }
            if (n == 0) {
                int row = zbase + v + rowlo;
                atomicAdd(&rowSl[row], S - de);     // diag-excluded row sum
                if (row < B_ROWS) posvl[row] = pl;  // pos logit (unique writer)
            }
        }
        // ---- col-side writeback: cols >= 128*(i+1) ----
        for (int j = 128 * (i + 1) + tid; j < NTOT; j += 256)
            atomicAdd(&rowSl[j], colacc[j]);
        __syncthreads();   // colacc reads done before next half re-inits
    }
}

// -------------------- finalize: sum (log S - pos)/N over rows --------------------
__global__ void finalize_kernel(const float* rowS, const float* posv, float* out) {
    int l = blockIdx.y;
    int r = blockIdx.x * 256 + threadIdx.x;
    float S  = rowS[(size_t)l * NTOT + r];
    float pl = posv[(size_t)l * B_ROWS + (r & (B_ROWS - 1))];  // symmetry for r>=2048
    float contrib = (__logf(S) - pl) * (1.0f / (float)NTOT);
    #pragma unroll
    for (int m = 16; m >= 1; m >>= 1) contrib += __shfl_xor(contrib, m);
    __shared__ float wacc[8];
    int wid = threadIdx.x >> 5, lane = threadIdx.x & 31;
    if (lane == 0) wacc[wid] = contrib;
    __syncthreads();
    if (threadIdx.x == 0) {
        float t = 0.f;
        #pragma unroll
        for (int w = 0; w < 8; ++w) t += wacc[w];
        atomicAdd(out, t);
    }
}

// -------------------- launch --------------------
extern "C" void kernel_launch(void* const* d_in, const int* in_sizes, int n_in,
                              void* d_out, int out_size, void* d_ws, size_t ws_size,
                              hipStream_t stream) {
    const float* f0 = (const float*)d_in[0];
    const float* f1 = (const float*)d_in[1];
    const float* f2 = (const float*)d_in[2];
    const float* f3 = (const float*)d_in[3];
    const float* f4 = (const float*)d_in[4];
    const float* f5 = (const float*)d_in[5];
    float* out = (float*)d_out;

    __bf16* zn  = (__bf16*)d_ws;                  // 24 MB normalized halves
    float* rowS = (float*)(zn + ZN_ELEMS);        // 9*4096 row sums
    float* posv = rowS + NLOSS * NTOT;            // 9*2048 positive logits

    zero_kernel<<<dim3((NLOSS * NTOT + 255) / 256), dim3(256), 0, stream>>>(out, rowS);
    row_normalize_kernel<<<dim3(6 * 2048), dim3(256), 0, stream>>>(
        f0, f1, f2, f3, f4, f5, zn);
    ortho_kernel<<<dim3(2048 / 8, 12), dim3(256), 0, stream>>>(
        f0, f1, f2, f3, f4, f5, out);
    ntxent_kernel<<<dim3(16, NLOSS), dim3(256), 0, stream>>>(zn, rowS, posv);
    finalize_kernel<<<dim3(NTOT / 256, NLOSS), dim3(256), 0, stream>>>(rowS, posv, out);
}
